// MultilayerMPNN_50749333569631
// MI455X (gfx1250) — compile-verified, matrix operand order checked by reading, not run on hardware
//
#include <hip/hip_runtime.h>
#include <hip/hip_bf16.h>

typedef __attribute__((ext_vector_type(16))) __bf16 v16bf;
typedef __attribute__((ext_vector_type(8)))  __bf16 v8bf;
typedef __attribute__((ext_vector_type(8)))  float  v8f;

#define EPB 128      // edges per block (4 waves x 32 rows)
#define ESTRIDE 168  // LDS row stride (bf16) for edge A-tiles; bank-conflict-free
#define NPB 128      // nodes per block (4 waves x 32 rows)
#define NSTRIDE 136  // LDS row stride (bf16) for node A-tiles; bank-conflict-free
#define BLK 128      // threads per block (4 waves)

// K-index map for the 16-bit A/B WMMA fragment layout (ISA 7.12.2):
// lanes 0-15 hold K = [0..7],[16..23]; lanes 16-31 hold K = [8..15],[24..31].
__device__ __forceinline__ int kmap(int i, int half) {
    return (i < 8) ? (8 * half + i) : (16 + 8 * half + (i - 8));
}

// ---- weight fragment prep: msg_W (3,136,64) -> bf16 frags [l][c=0..4][t=0..3][lane][16]
__global__ __launch_bounds__(256) void prep_msg_kernel(const float* __restrict__ mW,
                                                       __bf16* __restrict__ dstf) {
    int idx = blockIdx.x * 256 + threadIdx.x;
    if (idx >= 3 * 5 * 4 * 32 * 16) return;
    int i    = idx & 15;
    int lane = (idx >> 4) & 31;
    int t    = (idx >> 9) & 3;
    int c    = (idx >> 11) % 5;
    int l    = idx / (5 * 2048);
    int half = lane >> 4;
    int n    = t * 16 + (lane & 15);
    int K    = c * 32 + kmap(i, half);
    float v  = (K < 136) ? mW[((size_t)l * 136 + K) * 64 + n] : 0.0f;
    dstf[idx] = (__bf16)v;
}

// ---- out_W (3,128,64) -> bf16 frags [l][c=0..3][t=0..3][lane][16]
__global__ __launch_bounds__(256) void prep_out_kernel(const float* __restrict__ oW,
                                                       __bf16* __restrict__ dstf) {
    int idx = blockIdx.x * 256 + threadIdx.x;
    if (idx >= 3 * 4 * 4 * 32 * 16) return;
    int i    = idx & 15;
    int lane = (idx >> 4) & 31;
    int t    = (idx >> 9) & 3;
    int c    = (idx >> 11) & 3;
    int l    = idx >> 13;
    int half = lane >> 4;
    int n    = t * 16 + (lane & 15);
    int K    = c * 32 + kmap(i, half);
    float v  = oW[((size_t)l * 128 + K) * 64 + n];
    dstf[idx] = (__bf16)v;
}

__global__ __launch_bounds__(256) void zero_kernel(int* __restrict__ p, int n) {
    int i = blockIdx.x * 256 + threadIdx.x;
    if (i < n) p[i] = 0;
}

// ---- edge kernel: msg = relu([x_j, x_i, w] @ mW + mb); scatter-max into aggr[src]
// Each wave computes a 32-edge x 64-out tile: two 16-row A tiles share each B fragment.
__global__ __launch_bounds__(BLK) void mpnn_edge_kernel(
    const float* __restrict__ x, const int* __restrict__ src, const int* __restrict__ dst,
    const float* __restrict__ ew, const __bf16* __restrict__ wfrag,
    const float* __restrict__ bias, int* __restrict__ aggr, int E) {
    __shared__ __bf16 sA[EPB * ESTRIDE];
    __shared__ int sSrc[EPB], sDst[EPB];
    const int tid  = threadIdx.x;
    const int base = blockIdx.x * EPB;

    {
        int ge = base + tid;
        sSrc[tid] = (ge < E) ? src[ge] : 0;
        sDst[tid] = (ge < E) ? dst[ge] : 0;
    }
    __syncthreads();

    // stage [x_j(64) | x_i(64) | w(8) | pad(24)] per edge as bf16
    for (int i = tid; i < EPB * 160; i += BLK) {
        int e = i / 160;
        int k = i - e * 160;
        int ge = base + e;
        float v = 0.0f;
        if (ge < E) {
            if (k < 64)       v = x[(size_t)sDst[e] * 64 + k];
            else if (k < 128) v = x[(size_t)sSrc[e] * 64 + (k - 64)];
            else if (k < 136) v = ew[(size_t)ge * 8 + (k - 128)];
        }
        sA[e * ESTRIDE + k] = (__bf16)v;
    }
    __syncthreads();

    const int wave = tid >> 5, lane = tid & 31;
    const int half = lane >> 4, lp = lane & 15;
    const int rowBase = wave * 32;
    v8f accLo[4] = {};
    v8f accHi[4] = {};
    const __bf16* rp0 = &sA[(rowBase + lp) * ESTRIDE];
    const __bf16* rp1 = rp0 + 16 * ESTRIDE;

#pragma unroll
    for (int c = 0; c < 5; ++c) {
        int k0 = c * 32 + 8 * half;
        v8bf a0 = *(const v8bf*)(rp0 + k0);
        v8bf a1 = *(const v8bf*)(rp0 + k0 + 16);
        v16bf aLo = __builtin_shufflevector(a0, a1, 0,1,2,3,4,5,6,7,8,9,10,11,12,13,14,15);
        v8bf a2 = *(const v8bf*)(rp1 + k0);
        v8bf a3 = *(const v8bf*)(rp1 + k0 + 16);
        v16bf aHi = __builtin_shufflevector(a2, a3, 0,1,2,3,4,5,6,7,8,9,10,11,12,13,14,15);
        const __bf16* bbase = wfrag + (size_t)c * 4 * 512 + (size_t)lane * 16;
#pragma unroll
        for (int t = 0; t < 4; ++t) {
            v16bf b = *(const v16bf*)(bbase + (size_t)t * 512);
            accLo[t] = __builtin_amdgcn_wmma_f32_16x16x32_bf16(
                false, aLo, false, b, (short)0, accLo[t], false, false);
            accHi[t] = __builtin_amdgcn_wmma_f32_16x16x32_bf16(
                false, aHi, false, b, (short)0, accHi[t], false, false);
        }
    }

    // bias + relu + scatter-max (int atomicMax valid: values >= 0, aggr init 0)
#pragma unroll
    for (int t = 0; t < 4; ++t) {
        int n = t * 16 + lp;
        float bn = bias[n];
#pragma unroll
        for (int r = 0; r < 8; ++r) {
            int m0 = rowBase + r + 8 * half;
            int m1 = m0 + 16;
            if (base + m0 < E) {
                float v = accLo[t][r] + bn;
                v = v > 0.0f ? v : 0.0f;
                atomicMax(&aggr[(size_t)sSrc[m0] * 64 + n], __float_as_int(v));
            }
            if (base + m1 < E) {
                float v = accHi[t][r] + bn;
                v = v > 0.0f ? v : 0.0f;
                atomicMax(&aggr[(size_t)sSrc[m1] * 64 + n], __float_as_int(v));
            }
        }
    }
}

// ---- node kernel: out = relu([x, aggr] @ oW + ob); in-place safe (own rows only)
__global__ __launch_bounds__(BLK) void mpnn_node_kernel(
    const float* __restrict__ xin, const int* __restrict__ aggr,
    const __bf16* __restrict__ wfrag, const float* __restrict__ bias,
    float* __restrict__ xout, int NN) {
    __shared__ __bf16 sA[NPB * NSTRIDE];
    const int tid  = threadIdx.x;
    const int base = blockIdx.x * NPB;

    for (int i = tid; i < NPB * 128; i += BLK) {
        int e = i >> 7;
        int k = i & 127;
        int gn = base + e;
        float v = 0.0f;
        if (gn < NN)
            v = (k < 64) ? xin[(size_t)gn * 64 + k]
                         : __int_as_float(aggr[(size_t)gn * 64 + (k - 64)]);
        sA[e * NSTRIDE + k] = (__bf16)v;
    }
    __syncthreads();

    const int wave = tid >> 5, lane = tid & 31;
    const int half = lane >> 4, lp = lane & 15;
    const int rowBase = wave * 32;
    v8f accLo[4] = {};
    v8f accHi[4] = {};
    const __bf16* rp0 = &sA[(rowBase + lp) * NSTRIDE];
    const __bf16* rp1 = rp0 + 16 * NSTRIDE;

#pragma unroll
    for (int c = 0; c < 4; ++c) {
        int k0 = c * 32 + 8 * half;
        v8bf a0 = *(const v8bf*)(rp0 + k0);
        v8bf a1 = *(const v8bf*)(rp0 + k0 + 16);
        v16bf aLo = __builtin_shufflevector(a0, a1, 0,1,2,3,4,5,6,7,8,9,10,11,12,13,14,15);
        v8bf a2 = *(const v8bf*)(rp1 + k0);
        v8bf a3 = *(const v8bf*)(rp1 + k0 + 16);
        v16bf aHi = __builtin_shufflevector(a2, a3, 0,1,2,3,4,5,6,7,8,9,10,11,12,13,14,15);
        const __bf16* bbase = wfrag + (size_t)c * 4 * 512 + (size_t)lane * 16;
#pragma unroll
        for (int t = 0; t < 4; ++t) {
            v16bf b = *(const v16bf*)(bbase + (size_t)t * 512);
            accLo[t] = __builtin_amdgcn_wmma_f32_16x16x32_bf16(
                false, aLo, false, b, (short)0, accLo[t], false, false);
            accHi[t] = __builtin_amdgcn_wmma_f32_16x16x32_bf16(
                false, aHi, false, b, (short)0, accHi[t], false, false);
        }
    }

#pragma unroll
    for (int t = 0; t < 4; ++t) {
        int n = t * 16 + lp;
        float bn = bias[n];
#pragma unroll
        for (int r = 0; r < 8; ++r) {
            int m0 = rowBase + r + 8 * half;
            int m1 = m0 + 16;
            if (base + m0 < NN) {
                float v = accLo[t][r] + bn;
                xout[(size_t)(base + m0) * 64 + n] = v > 0.0f ? v : 0.0f;
            }
            if (base + m1 < NN) {
                float v = accHi[t][r] + bn;
                xout[(size_t)(base + m1) * 64 + n] = v > 0.0f ? v : 0.0f;
            }
        }
    }
}

extern "C" void kernel_launch(void* const* d_in, const int* in_sizes, int n_in,
                              void* d_out, int out_size, void* d_ws, size_t ws_size,
                              hipStream_t stream) {
    const float* x  = (const float*)d_in[0];
    const int*   ei = (const int*)d_in[1];
    const float* ew = (const float*)d_in[2];
    const float* mW = (const float*)d_in[3];
    const float* mb = (const float*)d_in[4];
    const float* oW = (const float*)d_in[5];
    const float* ob = (const float*)d_in[6];
    float* out = (float*)d_out;

    const int NN = in_sizes[0] / 64;
    const int E  = in_sizes[1] / 2;
    const int* src = ei;
    const int* dst = ei + E;

    char* ws = (char*)d_ws;
    int* aggr = (int*)ws;
    size_t aggrBytes = (size_t)NN * 64 * sizeof(int);
    size_t off = (aggrBytes + 255) & ~(size_t)255;
    const size_t msgElems = 3 * 5 * 4 * 32 * 16;  // 30720
    const size_t outElems = 3 * 4 * 4 * 32 * 16;  // 24576
    __bf16* msgFrag = (__bf16*)(ws + off);
    __bf16* outFrag = (__bf16*)(ws + off + msgElems * sizeof(__bf16));

    prep_msg_kernel<<<(int)((msgElems + 255) / 256), 256, 0, stream>>>(mW, msgFrag);
    prep_out_kernel<<<(int)((outElems + 255) / 256), 256, 0, stream>>>(oW, outFrag);

    const int gridE = (E + EPB - 1) / EPB;
    const int gridN = (NN + NPB - 1) / NPB;
    const int zeroGrid = (NN * 64 + 255) / 256;

    for (int l = 0; l < 3; ++l) {
        const float* xin = (l == 0) ? x : out;
        zero_kernel<<<zeroGrid, 256, 0, stream>>>(aggr, NN * 64);
        mpnn_edge_kernel<<<gridE, BLK, 0, stream>>>(
            xin, src, dst, ew, msgFrag + (size_t)l * 5 * 4 * 512, mb + l * 64, aggr, E);
        mpnn_node_kernel<<<gridN, BLK, 0, stream>>>(
            xin, aggr, outFrag + (size_t)l * 4 * 4 * 512, ob + l * 64, out, NN);
    }
}
